// NodeBlock_11373073400276
// MI455X (gfx1250) — compile-verified, matrix operand order checked by reading, not run on hardware
//
#include <hip/hip_runtime.h>

typedef __attribute__((ext_vector_type(16))) __bf16 v16bf;
typedef __attribute__((ext_vector_type(8)))  float  v8f;

#define N_NODES 50000
#define N_EDGES 600000
#define HPITCH  264          // padded LDS row pitch (bf16 elems) to avoid bank conflicts
#define WAVES   4            // waves per block in MLP kernel

// ---------------------------------------------------------------------------
// Pack f32 row-major weight W[K][Nw] into bf16 WMMA B-fragment tiles:
//   tileIdx = t*(K/32) + kk ; per tile: 32 lanes x 16 bf16 (contiguous).
// Lane layout (ISA 7.12.2, 16-bit B 32x16): lane = column n = t*16 + (lane&15),
//   element e -> k = kk*32 + (e&7) + (e>>3)*16 + (lane>>4)*8
// ---------------------------------------------------------------------------
__global__ void pack_weights(const float* __restrict__ W, __bf16* __restrict__ Wp,
                             int K, int Nw) {
    int tid = blockIdx.x * blockDim.x + threadIdx.x;
    int nK = K >> 5;                 // K/32
    int nT = Nw >> 4;                // Nw/16
    int total = nT * nK * 32;
    if (tid >= total) return;
    int lane    = tid & 31;
    int tileIdx = tid >> 5;          // t*nK + kk
    int kk = tileIdx % nK;
    int t  = tileIdx / nK;
    int half = lane >> 4, r = lane & 15;
    int n = t * 16 + r;
    __bf16* out = Wp + ((size_t)tileIdx << 9) + lane * 16;   // 512 bf16 per tile
    int k0 = kk * 32 + half * 8;
#pragma unroll
    for (int e = 0; e < 16; ++e) {
        int k = k0 + (e & 7) + ((e >> 3) * 16);
        out[e] = (__bf16)W[(size_t)k * Nw + n];
    }
}

__global__ void zero_f32(float* __restrict__ p, long n) {
    long i = (long)blockIdx.x * blockDim.x + threadIdx.x;
    if (i < n) p[i] = 0.0f;
}

// ---------------------------------------------------------------------------
// Scatter-add: 32 lanes per edge, each lane owns a float4 slice of the 128
// features. Coalesced 512B read per edge; native f32 atomics into agg which
// fits entirely in L2 (25.6MB << 192MB).
// ---------------------------------------------------------------------------
__global__ __launch_bounds__(256) void scatter_edges(const float* __restrict__ xe,
                                                     const long long* __restrict__ ei,
                                                     float* __restrict__ agg,
                                                     int nEdges) {
    int tid = blockIdx.x * 256 + threadIdx.x;
    int e = tid >> 5;
    if (e >= nEdges) return;
    int c = (tid & 31) * 4;
    const float4 v = *(const float4*)(xe + (size_t)e * 128 + c);
    long long s = ei[(size_t)e * 2 + 0];
    long long d = ei[(size_t)e * 2 + 1];
    float* ps = agg + (size_t)s * 128 + c;
    float* pd = agg + (size_t)d * 128 + c;
    unsafeAtomicAdd(ps + 0, v.x); unsafeAtomicAdd(ps + 1, v.y);
    unsafeAtomicAdd(ps + 2, v.z); unsafeAtomicAdd(ps + 3, v.w);
    unsafeAtomicAdd(pd + 0, v.x); unsafeAtomicAdd(pd + 1, v.y);
    unsafeAtomicAdd(pd + 2, v.z); unsafeAtomicAdd(pd + 3, v.w);
}

// ---------------------------------------------------------------------------
// Fused MLP: each wave owns a 16-node M-tile.
//   GEMM1: [16x256] @ [256x256] via 16 N-tiles x 8 K-steps of wmma bf16
//   relu+bias -> bf16 h staged in padded LDS -> rebuild A fragments
//   GEMM2: [16x256] @ [256x128] via 8 N-tiles x 8 K-steps
// A fragments for the whole K=256 are cached in registers (64 VGPRs).
// ---------------------------------------------------------------------------
__global__ __launch_bounds__(WAVES * 32) void mlp_kernel(
    const float* __restrict__ xnode, const float* __restrict__ agg,
    const __bf16* __restrict__ W1p, const __bf16* __restrict__ W2p,
    const float* __restrict__ b1, const float* __restrict__ b2,
    float* __restrict__ out, int nNodes)
{
    __shared__ __bf16 hst[WAVES][16 * HPITCH];

    const int wave = threadIdx.x >> 5;
    const int lane = threadIdx.x & 31;
    const int half = lane >> 4;
    const int r    = lane & 15;
    const int m0   = (blockIdx.x * WAVES + wave) * 16;

    // ---- load A fragments: row r of [x_node | agg], K = 0..255, bf16 ----
    int row = m0 + r;
    if (row >= nNodes) row = nNodes - 1;         // clamp tail (stores guarded)
    v16bf aF[8];
#pragma unroll
    for (int kk = 0; kk < 8; ++kk) {
        int c1 = kk * 32 + half * 8;             // first 8-elem chunk
        int c2 = c1 + 16;                        // second 8-elem chunk
        const float* s1 = (c1 < 128) ? (xnode + (size_t)row * 128 + c1)
                                     : (agg   + (size_t)row * 128 + (c1 - 128));
        const float* s2 = (c2 < 128) ? (xnode + (size_t)row * 128 + c2)
                                     : (agg   + (size_t)row * 128 + (c2 - 128));
        v16bf a;
#pragma unroll
        for (int j = 0; j < 8; ++j) a[j] = (__bf16)s1[j];
#pragma unroll
        for (int j = 0; j < 8; ++j) a[8 + j] = (__bf16)s2[j];
        aF[kk] = a;
    }

    // ---- GEMM1 + bias + relu -> LDS (bf16) ----
    __bf16* hrow = &hst[wave][0];
    for (int t = 0; t < 16; ++t) {
        v8f acc = {};
#pragma unroll
        for (int kk = 0; kk < 8; ++kk) {
            v16bf b = ((const v16bf*)(W1p + ((size_t)(t * 8 + kk) << 9)))[lane];
            acc = __builtin_amdgcn_wmma_f32_16x16x32_bf16(
                false, aF[kk], false, b, (short)0, acc, false, false);
        }
        float bias = b1[t * 16 + r];
#pragma unroll
        for (int i = 0; i < 8; ++i) {
            float v = acc[i] + bias;
            v = v > 0.0f ? v : 0.0f;
            hrow[(i + half * 8) * HPITCH + t * 16 + r] = (__bf16)v;
        }
    }
    __syncthreads();   // uniform control flow across all waves

    // ---- rebuild A fragments from h (LDS, padded pitch) ----
    v16bf hF[8];
#pragma unroll
    for (int kk = 0; kk < 8; ++kk) {
        int c1 = kk * 32 + half * 8;
        const __bf16* s1 = hrow + r * HPITCH + c1;
        v16bf a;
#pragma unroll
        for (int j = 0; j < 8; ++j) a[j] = s1[j];
#pragma unroll
        for (int j = 0; j < 8; ++j) a[8 + j] = s1[16 + j];
        hF[kk] = a;
    }

    // ---- GEMM2 + bias -> out ----
    for (int t = 0; t < 8; ++t) {
        v8f acc = {};
#pragma unroll
        for (int kk = 0; kk < 8; ++kk) {
            v16bf b = ((const v16bf*)(W2p + ((size_t)(t * 8 + kk) << 9)))[lane];
            acc = __builtin_amdgcn_wmma_f32_16x16x32_bf16(
                false, hF[kk], false, b, (short)0, acc, false, false);
        }
        float bias = b2[t * 16 + r];
#pragma unroll
        for (int i = 0; i < 8; ++i) {
            int orow = m0 + i + half * 8;
            if (orow < nNodes)
                out[(size_t)orow * 128 + t * 16 + r] = acc[i] + bias;
        }
    }
}

// ---------------------------------------------------------------------------
extern "C" void kernel_launch(void* const* d_in, const int* in_sizes, int n_in,
                              void* d_out, int out_size, void* d_ws, size_t ws_size,
                              hipStream_t stream) {
    (void)in_sizes; (void)n_in; (void)out_size; (void)ws_size;
    const float*     x_node = (const float*)d_in[0];
    const float*     x_edge = (const float*)d_in[1];
    const long long* ei     = (const long long*)d_in[2];   // int64 per reference
    const float*     W1     = (const float*)d_in[3];
    const float*     b1     = (const float*)d_in[4];
    const float*     W2     = (const float*)d_in[5];
    const float*     b2     = (const float*)d_in[6];

    // d_out layout (flat, return order): x_node_out f32 | x_edge f32 | edge_index i64
    float* node_out  = (float*)d_out;
    float* xedge_out = node_out + (size_t)N_NODES * 128;
    char*  ei_out    = (char*)(xedge_out + (size_t)N_EDGES * 128);

    // agg scratch aliases the x_edge output region; stream order makes this safe
    float* agg = xedge_out;

    // packed bf16 weights live in d_ws (~192 KB)
    __bf16* W1p = (__bf16*)d_ws;
    __bf16* W2p = W1p + 256 * 256;

    // 1) pack weights to bf16 WMMA layout (W1: 16*8*32=4096 thr, W2: 8*8*32=2048)
    pack_weights<<<(4096 + 255) / 256, 256, 0, stream>>>(W1, W1p, 256, 256);
    pack_weights<<<(2048 + 255) / 256, 256, 0, stream>>>(W2, W2p, 256, 128);

    // 2) zero agg, scatter edges (both endpoints) with native f32 atomics
    long aggN = (long)N_NODES * 128;
    zero_f32<<<(int)((aggN + 511) / 512), 512, 0, stream>>>(agg, aggN);
    long sthreads = (long)N_EDGES * 32;
    scatter_edges<<<(int)((sthreads + 255) / 256), 256, 0, stream>>>(x_edge, ei, agg, N_EDGES);

    // 3) fused WMMA MLP: 64 nodes per block (4 waves x 16-row M-tiles)
    mlp_kernel<<<(N_NODES + WAVES * 16 - 1) / (WAVES * 16), WAVES * 32, 0, stream>>>(
        x_node, agg, W1p, W2p, b1, b2, node_out, N_NODES);

    // 4) pass-through outputs (after MLP consumed agg)
    hipMemcpyAsync(xedge_out, x_edge, (size_t)N_EDGES * 128 * sizeof(float),
                   hipMemcpyDeviceToDevice, stream);
    hipMemcpyAsync(ei_out, ei, (size_t)N_EDGES * 2 * sizeof(long long),
                   hipMemcpyDeviceToDevice, stream);
}
